// Multihead_self_attention_73512660238389
// MI455X (gfx1250) — matrix operand order repaired and checked by reading, not compiled, not run on hardware
//
#include <hip/hip_runtime.h>
#include <hip/hip_bf16.h>

// ---------------------------------------------------------------------------
// Types for CDNA5 WMMA (wave32): v_wmma_f32_16x16x32_bf16
// ---------------------------------------------------------------------------
typedef __attribute__((ext_vector_type(16))) __bf16 bf16x16;
typedef __attribute__((ext_vector_type(8)))  float  f32x8;

#define HEADS    16
#define HEAD_DIM 64
#define DIM      1024
#define INNER    1024          // HEADS*HEAD_DIM
#define T3       3072          // 3*INNER
#define BATCH    2
#define SEQ      2048
#define ROWS     (BATCH*SEQ)   // 4096
#define LN_EPS   1e-5f

__device__ __forceinline__ f32x8 wmma_bf16(bf16x16 a, bf16x16 b, f32x8 c) {
  // (neg_a, A, neg_b, B, c_mod, C, reuse_a, reuse_b)
  return __builtin_amdgcn_wmma_f32_16x16x32_bf16(false, a, false, b, (short)0, c,
                                                 false, false);
}

// A-matrix 16x32 bf16 tile from row-major memory (row stride ldk elements).
// Lane l: m = l&15, k = (l>>4)*16 + j  ->  one contiguous
// 16-element (32B) run per lane.
__device__ __forceinline__ bf16x16 load_a_tile(const __bf16* __restrict__ base,
                                               int ldk, int lane) {
  const int m    = lane & 15;
  const int half = lane >> 4;
  const __bf16* p = base + (size_t)m * ldk + half * 16;
  union { uint4 u[2]; bf16x16 v; } t;
  t.u[0] = *(const uint4*)(p);        // k = half*16 .. half*16+7
  t.u[1] = *(const uint4*)(p + 8);    // k = half*16+8 .. half*16+15
  return t.v;
}

// B-matrix 32x16 bf16 tile, sourced from *column-major* (i.e. B^T row-major,
// row stride ldk elements). Lane l: n = l&15, k = (l>>4)*16 + j  ->  one
// contiguous 16-element (32B) run per lane.
__device__ __forceinline__ bf16x16 load_bT_tile(const __bf16* __restrict__ baseT,
                                                int ldk, int lane) {
  const int n    = lane & 15;
  const int half = lane >> 4;
  const __bf16* p = baseT + (size_t)n * ldk + half * 16;
  union { uint4 u[2]; bf16x16 v; } t;
  t.u[0] = *(const uint4*)(p);
  t.u[1] = *(const uint4*)(p + 8);
  return t.v;
}

// ---------------------------------------------------------------------------
// fp32 -> bf16 conversion kernels
// ---------------------------------------------------------------------------
__global__ __launch_bounds__(256) void cvt_bf16_kernel(const float* __restrict__ in,
                                                       __bf16* __restrict__ out,
                                                       int n) {
  int i = blockIdx.x * blockDim.x + threadIdx.x;
  if (i < n) out[i] = (__bf16)in[i];
}

// in: [rows][cols] fp32  ->  out: [cols][rows] bf16 (transposed)
__global__ __launch_bounds__(256) void cvtT_bf16_kernel(const float* __restrict__ in,
                                                        __bf16* __restrict__ out,
                                                        int rows, int cols) {
  int i = blockIdx.x * blockDim.x + threadIdx.x;
  if (i < rows * cols) {
    int c = i / rows;
    int r = i - c * rows;
    out[i] = (__bf16)in[(size_t)r * cols + c];
  }
}

// ---------------------------------------------------------------------------
// QKV projection: qkv[4096,3072] = x_bf16[4096,1024] @ w_qkv (via wqkvT[3072,1024])
// Block = 256 threads = 8 waves; each wave computes a 32x64 tile.
// Software-pipelined (double-buffered fragments); steady-state loop unrolled
// by 2 so the cur/nxt rotation is register renaming, not v_mov copies.
// grid = (ROWS/32, T3/512)
// ---------------------------------------------------------------------------
__global__ __launch_bounds__(256) void qkv_gemm_kernel(const __bf16* __restrict__ A,
                                                       const __bf16* __restrict__ BT,
                                                       __bf16* __restrict__ C) {
  const int lane = threadIdx.x & 31;
  const int wave = threadIdx.x >> 5;
  const int m0   = blockIdx.x * 32;
  const int n0   = blockIdx.y * 512 + wave * 64;
  const int K = DIM;

  f32x8 acc[2][4] = {};
  const __bf16* a_base = A + (size_t)m0 * K;

  // prologue: fragments for kc = 0
  bf16x16 a_cur[2], b_cur[4];
  a_cur[0] = load_a_tile(a_base, K, lane);
  a_cur[1] = load_a_tile(a_base + (size_t)16 * K, K, lane);
#pragma unroll
  for (int t = 0; t < 4; ++t)
    b_cur[t] = load_bT_tile(BT + (size_t)(n0 + t * 16) * K, K, lane);

  // steady state: load kc, consume kc-32
#pragma unroll 2
  for (int kc = 32; kc < K; kc += 32) {
    __builtin_prefetch(a_base + kc + 32, 0, 1);          // global_prefetch_b8
    bf16x16 a_nxt[2], b_nxt[4];
    a_nxt[0] = load_a_tile(a_base + kc, K, lane);
    a_nxt[1] = load_a_tile(a_base + (size_t)16 * K + kc, K, lane);
#pragma unroll
    for (int t = 0; t < 4; ++t)
      b_nxt[t] = load_bT_tile(BT + (size_t)(n0 + t * 16) * K + kc, K, lane);

#pragma unroll
    for (int t = 0; t < 4; ++t) {
      acc[0][t] = wmma_bf16(a_cur[0], b_cur[t], acc[0][t]);
      acc[1][t] = wmma_bf16(a_cur[1], b_cur[t], acc[1][t]);
    }
    a_cur[0] = a_nxt[0];
    a_cur[1] = a_nxt[1];
#pragma unroll
    for (int t = 0; t < 4; ++t) b_cur[t] = b_nxt[t];
  }
  // epilogue k-step
#pragma unroll
  for (int t = 0; t < 4; ++t) {
    acc[0][t] = wmma_bf16(a_cur[0], b_cur[t], acc[0][t]);
    acc[1][t] = wmma_bf16(a_cur[1], b_cur[t], acc[1][t]);
  }

  const int half = lane >> 4, nl = lane & 15;
#pragma unroll
  for (int mi = 0; mi < 2; ++mi)
#pragma unroll
    for (int t = 0; t < 4; ++t)
#pragma unroll
      for (int r = 0; r < 8; ++r)
        C[(size_t)(m0 + mi * 16 + half * 8 + r) * T3 + n0 + t * 16 + nl] =
            (__bf16)acc[mi][t][r];
}

// ---------------------------------------------------------------------------
// Flash-style attention. qkv bf16 [4096, 3072] -> attn bf16 [4096, 1024].
// Block = 256 threads = 8 waves; each wave owns 16 query rows of one (b,h).
// K-tiles double-buffered one 32-key chunk ahead; V gather issued before the
// S WMMAs so its latency hides under matrix + softmax work.
// grid = BATCH*HEADS*(SEQ/128) = 512
// ---------------------------------------------------------------------------
template <bool LOAD_NEXT>
__device__ __forceinline__ void attn_chunk(
    int c0, int lane, int half, int nl,
    const __bf16* __restrict__ kbase, const __bf16* __restrict__ vbase,
    bf16x16 qa0, bf16x16 qa1,
    bf16x16 kcur[4], bf16x16 knxt[4],
    f32x8 o[4], float rmax[8], float rsum[8],
    __bf16 (*plds)[32]) {
  const float scale = 0.125f;  // HEAD_DIM^-0.5

  // ---- V gather for this chunk (issued early; overlaps S math + softmax) ----
  bf16x16 vb[4];
#pragma unroll
  for (int t = 0; t < 4; ++t) {
    const __bf16* vp = vbase + (size_t)(c0 + half * 16) * T3 + t * 16 + nl;
#pragma unroll
    for (int j = 0; j < 16; ++j) vb[t][j] = vp[(size_t)j * T3];
  }

  // ---- S = (Q K^T) : two 16x16 f32 tiles over the 32-key chunk ----
  f32x8 s0 = {}, s1 = {};
  s0 = wmma_bf16(qa0, kcur[0], s0);
  s0 = wmma_bf16(qa1, kcur[1], s0);
  s1 = wmma_bf16(qa0, kcur[2], s1);
  s1 = wmma_bf16(qa1, kcur[3], s1);

  // ---- start next chunk's K-tile loads while softmax runs ----
  if (LOAD_NEXT) {
    const __bf16* kn = kbase + (size_t)(c0 + 32) * T3;
    __builtin_prefetch(kn + (size_t)32 * T3, 0, 1);
    knxt[0] = load_bT_tile(kn, T3, lane);
    knxt[1] = load_bT_tile(kn + 32, T3, lane);
    knxt[2] = load_bT_tile(kn + (size_t)16 * T3, T3, lane);
    knxt[3] = load_bT_tile(kn + (size_t)16 * T3 + 32, T3, lane);
  }

  // ---- online softmax update (rows in VGPR index, cols across 16 lanes) ----
#pragma unroll
  for (int r = 0; r < 8; ++r) {
    float a0 = s0[r] * scale;
    float a1 = s1[r] * scale;
    float cm = fmaxf(a0, a1);
    cm = fmaxf(cm, __shfl_xor(cm, 1, 16));
    cm = fmaxf(cm, __shfl_xor(cm, 2, 16));
    cm = fmaxf(cm, __shfl_xor(cm, 4, 16));
    cm = fmaxf(cm, __shfl_xor(cm, 8, 16));
    float nm    = fmaxf(rmax[r], cm);
    float alpha = __expf(rmax[r] - nm);
    float p0    = __expf(a0 - nm);
    float p1    = __expf(a1 - nm);
    plds[half * 8 + r][nl]      = (__bf16)p0;
    plds[half * 8 + r][16 + nl] = (__bf16)p1;
    float cs = p0 + p1;
    cs += __shfl_xor(cs, 1, 16);
    cs += __shfl_xor(cs, 2, 16);
    cs += __shfl_xor(cs, 4, 16);
    cs += __shfl_xor(cs, 8, 16);
    rsum[r] = rsum[r] * alpha + cs;
    rmax[r] = nm;
#pragma unroll
    for (int t = 0; t < 4; ++t) o[t][r] *= alpha;  // rescale O accumulators
  }

  // Wave-internal LDS handoff: C-layout P -> A-layout P (wave32 lockstep;
  // only need to drain this wave's DS ops, no workgroup barrier).
  __builtin_amdgcn_wave_barrier();
  asm volatile("s_wait_dscnt 0" ::: "memory");
  __builtin_amdgcn_wave_barrier();

  bf16x16 pa = load_a_tile(&plds[0][0], 32, lane);

  // ---- O += P @ V ----
#pragma unroll
  for (int t = 0; t < 4; ++t) o[t] = wmma_bf16(pa, vb[t], o[t]);
}

__global__ __launch_bounds__(256) void attention_kernel(const __bf16* __restrict__ qkv,
                                                        __bf16* __restrict__ attn) {
  const int lane = threadIdx.x & 31;
  const int wave = threadIdx.x >> 5;
  const int half = lane >> 4, nl = lane & 15;

  const int qblk = blockIdx.x & 15;          // SEQ/128 = 16 query blocks
  const int bh   = blockIdx.x >> 4;
  const int b    = bh >> 4;                  // HEADS = 16
  const int h    = bh & 15;
  const int q0   = qblk * 128 + wave * 16;

  const __bf16* qbase = qkv + (size_t)(b * SEQ + q0) * T3 + h * HEAD_DIM;
  const __bf16* kbase = qkv + (size_t)(b * SEQ) * T3 + INNER + h * HEAD_DIM;
  const __bf16* vbase = qkv + (size_t)(b * SEQ) * T3 + 2 * INNER + h * HEAD_DIM;

  // Q tile 16x64 in A-layout, resident in VGPRs for the whole kernel.
  bf16x16 qa0 = load_a_tile(qbase, T3, lane);       // d = 0..31
  bf16x16 qa1 = load_a_tile(qbase + 32, T3, lane);  // d = 32..63

  f32x8 o[4] = {};
  float rmax[8], rsum[8];
#pragma unroll
  for (int r = 0; r < 8; ++r) { rmax[r] = -1e30f; rsum[r] = 0.f; }

  __shared__ __align__(16) __bf16 plds[8][16][32];  // per-wave P staging

  // prologue: K tiles for chunk 0
  bf16x16 kcur[4], knxt[4];
  kcur[0] = load_bT_tile(kbase, T3, lane);
  kcur[1] = load_bT_tile(kbase + 32, T3, lane);
  kcur[2] = load_bT_tile(kbase + (size_t)16 * T3, T3, lane);
  kcur[3] = load_bT_tile(kbase + (size_t)16 * T3 + 32, T3, lane);

  int c0 = 0;
#pragma unroll 2
  for (; c0 < SEQ - 32; c0 += 32) {
    attn_chunk<true>(c0, lane, half, nl, kbase, vbase, qa0, qa1,
                     kcur, knxt, o, rmax, rsum, plds[wave]);
#pragma unroll
    for (int i = 0; i < 4; ++i) kcur[i] = knxt[i];
  }
  attn_chunk<false>(c0, lane, half, nl, kbase, vbase, qa0, qa1,
                    kcur, knxt, o, rmax, rsum, plds[wave]);

  // ---- epilogue: normalize by row sums, store bf16 [b, l, h*64 + d] ----
#pragma unroll
  for (int r = 0; r < 8; ++r) {
    const float inv  = 1.f / rsum[r];
    const size_t row = (size_t)(b * SEQ + q0 + half * 8 + r);
#pragma unroll
    for (int t = 0; t < 4; ++t)
      attn[row * INNER + h * HEAD_DIM + t * 16 + nl] = (__bf16)(o[t][r] * inv);
  }
}

// ---------------------------------------------------------------------------
// Output projection + bias + residual + LayerNorm, fused.
// Block = 256 threads = 8 waves; block owns 16 rows x 1024 cols
// (each wave a 16x128 slab), double-buffered k-loop (unrolled by 2 to avoid
// fragment-copy movs). grid = ROWS/16 = 256.
// ---------------------------------------------------------------------------
__global__ __launch_bounds__(256) void outproj_ln_kernel(
    const __bf16* __restrict__ A,    // attn bf16 [4096,1024]
    const __bf16* __restrict__ BT,   // woutT bf16 [1024,1024] (n-major)
    const float* __restrict__ bias,
    const float* __restrict__ x,     // residual fp32 [4096,1024]
    const float* __restrict__ gamma,
    const float* __restrict__ beta,
    float* __restrict__ out) {
  const int lane = threadIdx.x & 31;
  const int wave = threadIdx.x >> 5;
  const int half = lane >> 4, nl = lane & 15;
  const int m0   = blockIdx.x * 16;
  const int n0   = wave * 128;
  const int K = DIM, N = DIM;

  f32x8 acc[8] = {};
  const __bf16* a_base = A + (size_t)m0 * K;

  bf16x16 a_cur, b_cur[8];
  a_cur = load_a_tile(a_base, K, lane);
#pragma unroll
  for (int t = 0; t < 8; ++t)
    b_cur[t] = load_bT_tile(BT + (size_t)(n0 + t * 16) * K, K, lane);

#pragma unroll 2
  for (int kc = 32; kc < K; kc += 32) {
    __builtin_prefetch(a_base + kc + 32, 0, 1);
    bf16x16 a_nxt = load_a_tile(a_base + kc, K, lane);
    bf16x16 b_nxt[8];
#pragma unroll
    for (int t = 0; t < 8; ++t)
      b_nxt[t] = load_bT_tile(BT + (size_t)(n0 + t * 16) * K + kc, K, lane);

#pragma unroll
    for (int t = 0; t < 8; ++t) acc[t] = wmma_bf16(a_cur, b_cur[t], acc[t]);
    a_cur = a_nxt;
#pragma unroll
    for (int t = 0; t < 8; ++t) b_cur[t] = b_nxt[t];
  }
#pragma unroll
  for (int t = 0; t < 8; ++t) acc[t] = wmma_bf16(a_cur, b_cur[t], acc[t]);

  // bias + residual, in place in accumulators
#pragma unroll
  for (int t = 0; t < 8; ++t)
#pragma unroll
    for (int r = 0; r < 8; ++r) {
      const int row = half * 8 + r;
      const int col = n0 + t * 16 + nl;
      acc[t][r] += bias[col] + x[(size_t)(m0 + row) * N + col];
    }

  // per-row sum / sumsq: reduce over this wave's 128 cols (16 lanes per half)
  __shared__ float redS[8][16];
  __shared__ float redQ[8][16];
  float psum[8], psq[8];
#pragma unroll
  for (int r = 0; r < 8; ++r) {
    float s = 0.f, q = 0.f;
#pragma unroll
    for (int t = 0; t < 8; ++t) { float v = acc[t][r]; s += v; q += v * v; }
    s += __shfl_xor(s, 1, 16);  q += __shfl_xor(q, 1, 16);
    s += __shfl_xor(s, 2, 16);  q += __shfl_xor(q, 2, 16);
    s += __shfl_xor(s, 4, 16);  q += __shfl_xor(q, 4, 16);
    s += __shfl_xor(s, 8, 16);  q += __shfl_xor(q, 8, 16);
    psum[r] = s; psq[r] = q;
  }
  if (nl == 0) {
#pragma unroll
    for (int r = 0; r < 8; ++r) {
      redS[wave][half * 8 + r] = psum[r];
      redQ[wave][half * 8 + r] = psq[r];
    }
  }
  __syncthreads();

  // finish reduction across the 8 waves, normalize, write fp32 output
#pragma unroll
  for (int r = 0; r < 8; ++r) {
    const int row = half * 8 + r;
    float ts = 0.f, tq = 0.f;
#pragma unroll
    for (int w = 0; w < 8; ++w) { ts += redS[w][row]; tq += redQ[w][row]; }
    const float mean = ts * (1.f / (float)DIM);
    const float var  = tq * (1.f / (float)DIM) - mean * mean;
    const float inv  = rsqrtf(var + LN_EPS);
#pragma unroll
    for (int t = 0; t < 8; ++t) {
      const int col = n0 + t * 16 + nl;
      out[(size_t)(m0 + row) * N + col] =
          (acc[t][r] - mean) * inv * gamma[col] + beta[col];
    }
  }
}

// ---------------------------------------------------------------------------
// Host launcher
// ---------------------------------------------------------------------------
extern "C" void kernel_launch(void* const* d_in, const int* in_sizes, int n_in,
                              void* d_out, int out_size, void* d_ws, size_t ws_size,
                              hipStream_t stream) {
  const float* x      = (const float*)d_in[0];
  const float* w_qkv  = (const float*)d_in[1];
  const float* w_out  = (const float*)d_in[2];
  const float* b_out  = (const float*)d_in[3];
  const float* gamma  = (const float*)d_in[4];
  const float* beta   = (const float*)d_in[5];
  float* out = (float*)d_out;

  char* ws = (char*)d_ws;
  __bf16* x_bf   = (__bf16*)ws;  ws += (size_t)ROWS * DIM * 2;     //  8 MB
  __bf16* wqkvT  = (__bf16*)ws;  ws += (size_t)T3 * DIM * 2;       //  6 MB
  __bf16* woutT  = (__bf16*)ws;  ws += (size_t)DIM * DIM * 2;      //  2 MB
  __bf16* qkv    = (__bf16*)ws;  ws += (size_t)ROWS * T3 * 2;      // 24 MB
  __bf16* attn   = (__bf16*)ws;  ws += (size_t)ROWS * INNER * 2;   //  8 MB

  {
    int n = ROWS * DIM;
    cvt_bf16_kernel<<<(n + 255) / 256, 256, 0, stream>>>(x, x_bf, n);
  }
  {
    int n = DIM * T3;
    cvtT_bf16_kernel<<<(n + 255) / 256, 256, 0, stream>>>(w_qkv, wqkvT, DIM, T3);
  }
  {
    int n = DIM * DIM;
    cvtT_bf16_kernel<<<(n + 255) / 256, 256, 0, stream>>>(w_out, woutT, DIM, DIM);
  }

  qkv_gemm_kernel<<<dim3(ROWS / 32, T3 / 512), 256, 0, stream>>>(x_bf, wqkvT, qkv);
  attention_kernel<<<BATCH * HEADS * (SEQ / 128), 256, 0, stream>>>(qkv, attn);
  outproj_ln_kernel<<<ROWS / 16, 256, 0, stream>>>(attn, woutT, b_out, x,
                                                   gamma, beta, out);
}
